// EncoderBlock_8005819040402
// MI455X (gfx1250) — matrix operand order repaired and checked by reading, not compiled, not run on hardware
//
#include <hip/hip_runtime.h>
#include <hip/hip_bf16.h>

// ---------------------------------------------------------------------------
// Encoder block on MI455X (gfx1250, wave32, WMMA).
// Math path: f16 operands + f32 WMMA accumulation (v_wmma_f32_16x16x32_f16).
// fp32 for layernorm / softmax / residual adds / final output.
// Global->LDS staging uses gfx1250 async-to-LDS DMA (ASYNCcnt) with double
// buffering; guarded by __has_builtin so the file compiles on any toolchain.
// ---------------------------------------------------------------------------

typedef _Float16 half_t;
typedef __attribute__((ext_vector_type(16))) _Float16 v16h;
typedef __attribute__((ext_vector_type(8)))  _Float16 v8h;
typedef __attribute__((ext_vector_type(4)))  _Float16 v4h;
typedef __attribute__((ext_vector_type(8)))  float    v8f;
typedef __attribute__((ext_vector_type(4)))  float    v4f;
typedef __attribute__((ext_vector_type(4)))  int      v4i;

#define CAT16(lo, hi) __builtin_shufflevector(lo, hi, 0,1,2,3,4,5,6,7,8,9,10,11,12,13,14,15)
#define AS1 __attribute__((address_space(1)))
#define AS3 __attribute__((address_space(3)))

#if defined(__AMDGCN__) && __has_builtin(__builtin_amdgcn_global_load_async_to_lds_b128) && \
    __has_builtin(__builtin_amdgcn_s_wait_asynccnt)
#define USE_ASYNC_LDS 1
#else
#define USE_ASYNC_LDS 0
#endif

// Copy 16 bytes global -> LDS. Async DMA (ASYNCcnt) on gfx1250 when available.
__device__ __forceinline__ void copy16_g2l(const half_t* g, half_t* l) {
#if USE_ASYNC_LDS
  __builtin_amdgcn_global_load_async_to_lds_b128((AS1 v4i*)g, (AS3 v4i*)l, 0, 0);
#else
  *(v8h*)l = *(const v8h*)g;
#endif
}
__device__ __forceinline__ void g2l_wait() {
#if USE_ASYNC_LDS
  __builtin_amdgcn_s_wait_asynccnt(0);
#endif
}

static constexpr int Bb   = 8;
static constexpr int Ss   = 1024;
static constexpr int Dd   = 1024;
static constexpr int Hh   = 16;
static constexpr int DKk  = 64;
static constexpr int DFFf = 4096;
static constexpr int NTOK = Bb * Ss;   // 8192

static constexpr int BM  = 128;
static constexpr int BN  = 128;
static constexpr int BK  = 64;
static constexpr int PAD = 8;          // halfs of padding per LDS row
static constexpr int TPB = 128;        // 4 waves of 32

// ---------------------------------------------------------------------------
// One-shot f32 -> f16 weight conversion (pure bandwidth, ~24 MB total).
// ---------------------------------------------------------------------------
__global__ __launch_bounds__(256)
void cvt_f32_f16_kernel(const float* __restrict__ src, half_t* __restrict__ dst, int n4) {
  int i = blockIdx.x * 256 + threadIdx.x;
  if (i < n4) {
    v4f w = *(const v4f*)(src + (size_t)i * 4);
    v4h h = {(_Float16)w[0], (_Float16)w[1], (_Float16)w[2], (_Float16)w[3]};
    *(v4h*)(dst + (size_t)i * 4) = h;
  }
}

// ---------------------------------------------------------------------------
// LayerNorm (fp32 in, f16 out). Scalar gamma/beta, unbiased variance (ddof=1)
// to match the reference. One block (256 threads) per 1024-element row.
// ---------------------------------------------------------------------------
__global__ __launch_bounds__(256)
void ln_f16_kernel(const float* __restrict__ x, const float* __restrict__ g,
                   const float* __restrict__ bt, half_t* __restrict__ out) {
  const size_t row = blockIdx.x;
  const float* xr = x + row * Dd;
  float vals[4];
  float s = 0.f, ss = 0.f;
  #pragma unroll
  for (int i = 0; i < 4; ++i) {
    float v = xr[threadIdx.x + i * 256];
    vals[i] = v; s += v; ss += v * v;
  }
  #pragma unroll
  for (int off = 1; off < 32; off <<= 1) {
    s  += __shfl_xor(s, off, 32);
    ss += __shfl_xor(ss, off, 32);
  }
  __shared__ float rs[8], rss[8];
  const int wv = threadIdx.x >> 5, ln = threadIdx.x & 31;
  if (ln == 0) { rs[wv] = s; rss[wv] = ss; }
  __syncthreads();
  if (wv == 0) {
    s  = (ln < 8) ? rs[ln]  : 0.f;
    ss = (ln < 8) ? rss[ln] : 0.f;
    #pragma unroll
    for (int off = 1; off < 8; off <<= 1) {
      s  += __shfl_xor(s, off, 32);
      ss += __shfl_xor(ss, off, 32);
    }
    if (ln == 0) { rs[0] = s; rss[0] = ss; }
  }
  __syncthreads();
  s = rs[0]; ss = rss[0];
  const float mu  = s * (1.0f / Dd);
  const float var = (ss - Dd * mu * mu) * (1.0f / (Dd - 1));   // ddof=1
  const float inv = rsqrtf(var + 1e-5f);
  const float g0 = g[0], b0 = bt[0];
  half_t* orow = out + row * Dd;
  #pragma unroll
  for (int i = 0; i < 4; ++i)
    orow[threadIdx.x + i * 256] = (half_t)(g0 * (vals[i] - mu) * inv + b0);
}

// ---------------------------------------------------------------------------
// GEMM: C[M,N] = A[M,K](f16) * W[N,K](f16)^T + bias (+resid f32) (+ReLU)
// Block tile 128x128, 4 waves each computing 64x64 via 16 wmma per 32-k step.
// Double-buffered LDS staging; tiles are pure 16B copies -> async-to-LDS DMA.
// ---------------------------------------------------------------------------
template <typename OutT, bool RELU, bool RESID>
__global__ __launch_bounds__(TPB, 1)
void gemm_wmma_kernel(const half_t* __restrict__ A, const half_t* __restrict__ W,
                      const float* __restrict__ bias, const float* __restrict__ resid,
                      OutT* __restrict__ C, int M, int N, int K) {
  __shared__ half_t As[2][BM][BK + PAD];
  __shared__ half_t Bs[2][BN][BK + PAD];

  const int t     = threadIdx.x;
  const int lane  = t & 31;
  const int wave  = t >> 5;
  const int wr    = (wave >> 1) * 64;      // wave row origin in block tile
  const int wc    = (wave & 1) * 64;       // wave col origin
  const int lmod  = lane & 15;
  const int lhalf = lane >> 4;
  const int bm = blockIdx.y * BM;
  const int bn = blockIdx.x * BN;

  // staging coordinates for this thread: 8 x 16B chunks per tile
  const int sr = t >> 3;                   // rows t/8 stepping by 16
  const int scol = (t & 7) << 3;           // 8-half column chunk

  auto stage = [&](int buf, int k0) {
    #pragma unroll
    for (int i = 0; i < 8; ++i) {
      int r = sr + i * 16;
      copy16_g2l(A + (size_t)(bm + r) * K + k0 + scol, &As[buf][r][scol]);
      copy16_g2l(W + (size_t)(bn + r) * K + k0 + scol, &Bs[buf][r][scol]);
    }
  };

  const v8f vzero = {0.f,0.f,0.f,0.f,0.f,0.f,0.f,0.f};
  v8f acc[4][4];
  #pragma unroll
  for (int i = 0; i < 4; ++i)
    #pragma unroll
    for (int j = 0; j < 4; ++j) acc[i][j] = vzero;

  stage(0, 0);
  g2l_wait();
  __syncthreads();

  int cur = 0;
  for (int k0 = 0; k0 < K; k0 += BK) {
    if (k0 + BK < K) stage(cur ^ 1, k0 + BK);   // overlap next-tile DMA

    #pragma unroll
    for (int ks = 0; ks < BK; ks += 32) {
      // A fragments: 16x32 f16, lane holds row m=lmod,
      //   e<8 -> k = ks + 8*lhalf + e ; e>=8 -> k = ks + 16 + 8*lhalf + (e-8)
      v16h af[4];
      #pragma unroll
      for (int mi = 0; mi < 4; ++mi) {
        int r = wr + mi * 16 + lmod;
        v8h lo = *(const v8h*)&As[cur][r][ks + lhalf * 8];
        v8h hi = *(const v8h*)&As[cur][r][ks + 16 + lhalf * 8];
        af[mi] = CAT16(lo, hi);
      }
      #pragma unroll
      for (int nj = 0; nj < 4; ++nj) {
        // B fragment: 32x16 f16, lane holds col n=lmod, same k order as A
        int r = wc + nj * 16 + lmod;
        v8h lo = *(const v8h*)&Bs[cur][r][ks + lhalf * 8];
        v8h hi = *(const v8h*)&Bs[cur][r][ks + 16 + lhalf * 8];
        v16h bf = CAT16(lo, hi);
        #pragma unroll
        for (int mi = 0; mi < 4; ++mi)
          acc[mi][nj] = __builtin_amdgcn_wmma_f32_16x16x32_f16(
              false, af[mi], false, bf, (short)0, acc[mi][nj], false, false);
      }
    }
    g2l_wait();
    __syncthreads();
    cur ^= 1;
  }

  // ---- epilogue: C layout is m = vg + 8*lhalf, n = lmod ----
  #pragma unroll
  for (int mi = 0; mi < 4; ++mi) {
    #pragma unroll
    for (int nj = 0; nj < 4; ++nj) {
      int col = bn + wc + nj * 16 + lmod;
      float bv = bias[col];
      #pragma unroll
      for (int vg = 0; vg < 8; ++vg) {
        int row = bm + wr + mi * 16 + lhalf * 8 + vg;
        float val = acc[mi][nj][vg] + bv;
        if (RESID) val += resid[(size_t)row * N + col];
        if (RELU)  val = fmaxf(val, 0.f);
        C[(size_t)row * N + col] = (OutT)val;
      }
    }
  }
}

// ---------------------------------------------------------------------------
// Flash attention: softmax(Q K^T / 8) V per (b,h). 64 query rows per block,
// one wave per 16 query rows, streaming 64-key chunks through LDS.
// Mask is a no-op in the reference (result of masked_fill discarded).
// ---------------------------------------------------------------------------
__global__ __launch_bounds__(TPB, 1)
void attn_flash_kernel(const half_t* __restrict__ q, const half_t* __restrict__ k,
                       const half_t* __restrict__ v, half_t* __restrict__ o) {
  __shared__ half_t Ks[64][DKk + PAD];
  __shared__ half_t Vt[DKk][64 + PAD];          // transposed: [dk][key]
  __shared__ half_t Ps[4][16][64 + PAD];        // per-wave P staging

  const int t     = threadIdx.x;
  const int lane  = t & 31;
  const int wave  = t >> 5;
  const int lmod  = lane & 15;
  const int lhalf = lane >> 4;

  const int bh = blockIdx.y;                    // b*H + h
  const int b  = bh / Hh, h = bh % Hh;
  const int q0 = blockIdx.x * 64;
  const size_t headoff = (size_t)b * Ss * Dd + (size_t)h * DKk;

  // Q fragments for this wave's 16 rows (row m = lmod), dk split into 2x32
  const half_t* Qr = q + headoff + (size_t)(q0 + wave * 16 + lmod) * Dd;
  v16h qf[2];
  #pragma unroll
  for (int kk = 0; kk < 2; ++kk) {
    v8h lo = *(const v8h*)(Qr + kk * 32 + lhalf * 8);
    v8h hi = *(const v8h*)(Qr + kk * 32 + 16 + lhalf * 8);
    qf[kk] = CAT16(lo, hi);
  }

  const v8f vzero = {0.f,0.f,0.f,0.f,0.f,0.f,0.f,0.f};
  v8f oacc[4];
  #pragma unroll
  for (int i = 0; i < 4; ++i) oacc[i] = vzero;
  float mSt[8], lSt[8];
  #pragma unroll
  for (int i = 0; i < 8; ++i) { mSt[i] = -1e30f; lSt[i] = 0.f; }

  for (int c0 = 0; c0 < Ss; c0 += 64) {
    // stage K rows [key][dk] (async DMA) and V transposed [dk][key] (manual)
    #pragma unroll
    for (int i = 0; i < 4; ++i) {
      int idx = i * TPB + t;
      int r = idx >> 3, cc = (idx & 7) << 3;
      copy16_g2l(k + headoff + (size_t)(c0 + r) * Dd + cc, &Ks[r][cc]);
      v8h vv = *(const v8h*)(v + headoff + (size_t)(c0 + r) * Dd + cc);
      #pragma unroll
      for (int e = 0; e < 8; ++e) Vt[cc + e][r] = vv[e];
    }
    g2l_wait();
    __syncthreads();

    // S = Q K^T (4 fragments of 16 keys, reduce over dk=64 in 2 wmma)
    v8f sc[4];
    #pragma unroll
    for (int nj = 0; nj < 4; ++nj) {
      sc[nj] = vzero;
      int kr = nj * 16 + lmod;                   // key column for B fragment
      #pragma unroll
      for (int kk = 0; kk < 2; ++kk) {
        v8h lo = *(const v8h*)&Ks[kr][kk * 32 + lhalf * 8];
        v8h hi = *(const v8h*)&Ks[kr][kk * 32 + 16 + lhalf * 8];
        v16h bf = CAT16(lo, hi);
        sc[nj] = __builtin_amdgcn_wmma_f32_16x16x32_f16(
            false, qf[kk], false, bf, (short)0, sc[nj], false, false);
      }
    }

    // online softmax update, per accumulator row (m = vg + 8*lhalf)
    #pragma unroll
    for (int vg = 0; vg < 8; ++vg) {
      float mloc = -1e30f;
      #pragma unroll
      for (int nj = 0; nj < 4; ++nj) {
        sc[nj][vg] *= 0.125f;                    // 1/sqrt(DK)
        mloc = fmaxf(mloc, sc[nj][vg]);
      }
      #pragma unroll
      for (int off = 1; off < 16; off <<= 1)
        mloc = fmaxf(mloc, __shfl_xor(mloc, off, 32));
      float mnew = fmaxf(mSt[vg], mloc);
      float corr = __expf(mSt[vg] - mnew);
      mSt[vg] = mnew;
      float rsum = 0.f;
      int prow = vg + lhalf * 8;
      #pragma unroll
      for (int nj = 0; nj < 4; ++nj) {
        float p = __expf(sc[nj][vg] - mnew);
        rsum += p;
        Ps[wave][prow][nj * 16 + lmod] = (half_t)p;
      }
      #pragma unroll
      for (int off = 1; off < 16; off <<= 1)
        rsum += __shfl_xor(rsum, off, 32);
      lSt[vg] = lSt[vg] * corr + rsum;
      #pragma unroll
      for (int dj = 0; dj < 4; ++dj) oacc[dj][vg] *= corr;
    }
    __syncthreads();                              // P visible across lanes

    // O += P V  (A = P 16x64 over keys, B = V 64x64 via Vt)
    v16h pf[2];
    #pragma unroll
    for (int kk = 0; kk < 2; ++kk) {
      v8h lo = *(const v8h*)&Ps[wave][lmod][kk * 32 + lhalf * 8];
      v8h hi = *(const v8h*)&Ps[wave][lmod][kk * 32 + 16 + lhalf * 8];
      pf[kk] = CAT16(lo, hi);
    }
    #pragma unroll
    for (int dj = 0; dj < 4; ++dj) {
      int dr = dj * 16 + lmod;                    // dk column
      #pragma unroll
      for (int kk = 0; kk < 2; ++kk) {
        v8h lo = *(const v8h*)&Vt[dr][kk * 32 + lhalf * 8];
        v8h hi = *(const v8h*)&Vt[dr][kk * 32 + 16 + lhalf * 8];
        v16h bf = CAT16(lo, hi);
        oacc[dj] = __builtin_amdgcn_wmma_f32_16x16x32_f16(
            false, pf[kk], false, bf, (short)0, oacc[dj], false, false);
      }
    }
    __syncthreads();                              // before K/V/P re-staging
  }

  // write O / l
  #pragma unroll
  for (int dj = 0; dj < 4; ++dj) {
    #pragma unroll
    for (int vg = 0; vg < 8; ++vg) {
      int row = q0 + wave * 16 + lhalf * 8 + vg;
      int col = dj * 16 + lmod;
      float val = oacc[dj][vg] / lSt[vg];
      o[headoff + (size_t)row * Dd + col] = (half_t)val;
    }
  }
}

// ---------------------------------------------------------------------------
// Launcher. Workspace layout (bytes), reusing dead buffers:
//   q:0 k:16M v:32M h1/attn:48M x2(f32):64M h2:96M f1:112M..176M
//   f16 weights: wq:176M wk:178M wv:180M wo:182M w1:184M w2:192M (end 200M)
// ---------------------------------------------------------------------------
static constexpr size_t MB = 1024ull * 1024ull;
static constexpr size_t OFF_Q   = 0;
static constexpr size_t OFF_K   = 16 * MB;
static constexpr size_t OFF_V   = 32 * MB;
static constexpr size_t OFF_H1  = 48 * MB;   // reused as attention output
static constexpr size_t OFF_X2  = 64 * MB;   // fp32
static constexpr size_t OFF_H2  = 96 * MB;
static constexpr size_t OFF_F1  = 112 * MB;
static constexpr size_t OFF_WQ  = 176 * MB;
static constexpr size_t OFF_WK  = 178 * MB;
static constexpr size_t OFF_WV  = 180 * MB;
static constexpr size_t OFF_WO  = 182 * MB;
static constexpr size_t OFF_W1  = 184 * MB;
static constexpr size_t OFF_W2  = 192 * MB;

extern "C" void kernel_launch(void* const* d_in, const int* in_sizes, int n_in,
                              void* d_out, int out_size, void* d_ws, size_t ws_size,
                              hipStream_t stream) {
  (void)in_sizes; (void)n_in; (void)out_size; (void)ws_size;
  const float* x   = (const float*)d_in[0];
  // d_in[1] = src_mask: no-op in the reference math
  const float* wq  = (const float*)d_in[2];
  const float* bq  = (const float*)d_in[3];
  const float* wk  = (const float*)d_in[4];
  const float* bk  = (const float*)d_in[5];
  const float* wv  = (const float*)d_in[6];
  const float* bv  = (const float*)d_in[7];
  const float* wo  = (const float*)d_in[8];
  const float* bo  = (const float*)d_in[9];
  const float* w1  = (const float*)d_in[10];
  const float* b1  = (const float*)d_in[11];
  const float* w2  = (const float*)d_in[12];
  const float* b2  = (const float*)d_in[13];
  const float* g1  = (const float*)d_in[14];
  const float* be1 = (const float*)d_in[15];
  const float* g2  = (const float*)d_in[16];
  const float* be2 = (const float*)d_in[17];
  float* out = (float*)d_out;

  char* ws = (char*)d_ws;
  half_t* qb   = (half_t*)(ws + OFF_Q);
  half_t* kb   = (half_t*)(ws + OFF_K);
  half_t* vb   = (half_t*)(ws + OFF_V);
  half_t* h1   = (half_t*)(ws + OFF_H1);
  half_t* attn = (half_t*)(ws + OFF_H1);   // h1 dead after QKV GEMMs
  float*  x2   = (float*)(ws + OFF_X2);
  half_t* h2   = (half_t*)(ws + OFF_H2);
  half_t* f1   = (half_t*)(ws + OFF_F1);
  half_t* wqh  = (half_t*)(ws + OFF_WQ);
  half_t* wkh  = (half_t*)(ws + OFF_WK);
  half_t* wvh  = (half_t*)(ws + OFF_WV);
  half_t* woh  = (half_t*)(ws + OFF_WO);
  half_t* w1h  = (half_t*)(ws + OFF_W1);
  half_t* w2h  = (half_t*)(ws + OFF_W2);

  const dim3 blk(TPB);
  const dim3 gD(Dd / BN, NTOK / BM);       // (8, 64)
  const dim3 gF(DFFf / BN, NTOK / BM);     // (32, 64)

  // 0) weights -> f16 staging (once per launch; pure bandwidth)
  const int nDD4  = Dd * Dd / 4;           // 262144
  const int nFD4  = DFFf * Dd / 4;         // 1048576
  cvt_f32_f16_kernel<<<(nDD4 + 255) / 256, 256, 0, stream>>>(wq, wqh, nDD4);
  cvt_f32_f16_kernel<<<(nDD4 + 255) / 256, 256, 0, stream>>>(wk, wkh, nDD4);
  cvt_f32_f16_kernel<<<(nDD4 + 255) / 256, 256, 0, stream>>>(wv, wvh, nDD4);
  cvt_f32_f16_kernel<<<(nDD4 + 255) / 256, 256, 0, stream>>>(wo, woh, nDD4);
  cvt_f32_f16_kernel<<<(nFD4 + 255) / 256, 256, 0, stream>>>(w1, w1h, nFD4);
  cvt_f32_f16_kernel<<<(nFD4 + 255) / 256, 256, 0, stream>>>(w2, w2h, nFD4);

  // 1) h1 = LN(x)
  ln_f16_kernel<<<NTOK, 256, 0, stream>>>(x, g1, be1, h1);
  // 2-4) q,k,v projections
  gemm_wmma_kernel<half_t, false, false><<<gD, blk, 0, stream>>>(h1, wqh, bq, nullptr, qb, NTOK, Dd, Dd);
  gemm_wmma_kernel<half_t, false, false><<<gD, blk, 0, stream>>>(h1, wkh, bk, nullptr, kb, NTOK, Dd, Dd);
  gemm_wmma_kernel<half_t, false, false><<<gD, blk, 0, stream>>>(h1, wvh, bv, nullptr, vb, NTOK, Dd, Dd);
  // 5) flash attention
  attn_flash_kernel<<<dim3(Ss / 64, Bb * Hh), blk, 0, stream>>>(qb, kb, vb, attn);
  // 6) x2 = x + attn @ wo^T + bo
  gemm_wmma_kernel<float, false, true><<<gD, blk, 0, stream>>>(attn, woh, bo, x, x2, NTOK, Dd, Dd);
  // 7) h2 = LN(x2)
  ln_f16_kernel<<<NTOK, 256, 0, stream>>>(x2, g2, be2, h2);
  // 8) f1 = relu(h2 @ w1^T + b1)
  gemm_wmma_kernel<half_t, true, false><<<gF, blk, 0, stream>>>(h2, w1h, b1, nullptr, f1, NTOK, DFFf, Dd);
  // 9) out = x2 + f1 @ w2^T + b2
  gemm_wmma_kernel<float, false, true><<<gD, blk, 0, stream>>>(f1, w2h, b2, x2, out, NTOK, Dd, DFFf);
}
